// EMAU_ORG_43834436223341
// MI455X (gfx1250) — compile-verified, matrix operand order checked by reading, not run on hardware
//
#include <hip/hip_runtime.h>
#include <hip/hip_bf16.h>

#define BB 16
#define CC 512
#define NN 4096   // H*W
#define KK 64
#define BN_EPS_F 1e-5f
#define CSTRIDE 20  // uint32 column stride (80B: 16B-aligned, bank-friendly)

typedef __bf16 bf16_t;
typedef __attribute__((ext_vector_type(16))) __bf16 v16bf;
typedef __attribute__((ext_vector_type(8)))  float  v8f;

static __device__ __forceinline__ v8f wmma_bf16(v16bf a, v16bf b, v8f c) {
  // D(f32 16x16) = A(bf16 16x32) * B(bf16 32x16) + C
  return __builtin_amdgcn_wmma_f32_16x16x32_bf16(false, a, false, b, (short)0, c,
                                                 false, false);
}

// pack two floats into one dword of bf16 pair (K, K+1) — matches WMMA VGPR layout
static __device__ __forceinline__ unsigned pack2bf(float lo, float hi) {
  unsigned short l = __builtin_bit_cast(unsigned short, (__bf16)lo);
  unsigned short h = __builtin_bit_cast(unsigned short, (__bf16)hi);
  return (unsigned)l | ((unsigned)h << 16);
}

// fragment from column-major packed LDS tile: two b128 reads, zero repacking
static __device__ __forceinline__ v16bf frag_lds(const unsigned* ldsU, int col, int hi) {
  union { v16bf v; int4 q[2]; } f;
  const int4* p = (const int4*)(ldsU + col * CSTRIDE + hi * 8);
  f.q[0] = p[0];
  f.q[1] = p[1];
  return f.v;
}

// fragment from global row (16 contiguous bf16): two global b128 reads
static __device__ __forceinline__ v16bf frag_global(const bf16_t* p) {
  union { v16bf v; int4 q[2]; } f;
  f.q[0] = ((const int4*)p)[0];
  f.q[1] = ((const int4*)p)[1];
  return f.v;
}

// ---------------------------------------------------------------- utilities
__global__ void zero_kernel(float* __restrict__ p, int n) {
  int i = blockIdx.x * 256 + threadIdx.x;
  if (i < n) p[i] = 0.f;
}

__global__ void cvt_weights_kernel(const float* __restrict__ w1,
                                   const float* __restrict__ w2,
                                   const float* __restrict__ mu_in,
                                   bf16_t* __restrict__ w1b,
                                   bf16_t* __restrict__ w2b,
                                   bf16_t* __restrict__ mub) {
  int i = blockIdx.x * 256 + threadIdx.x;
  if (i < CC * CC) { w1b[i] = (bf16_t)w1[i]; w2b[i] = (bf16_t)w2[i]; }
  if (i < CC * KK) {
    bf16_t v = (bf16_t)mu_in[i];
    #pragma unroll
    for (int b = 0; b < BB; ++b) mub[(size_t)b * CC * KK + i] = v;
  }
}

// ---------------------------------------------------------------- conv1
// xf[b,o,n] = bf16( sum_c w1[o,c]*x[b,c,n] + b1[o] )
// grid (NN/64, CC/128, BB), block 256 (8 waves); wave = 16(o) x 64(n)
__global__ void __launch_bounds__(256)
conv1_kernel(const float* __restrict__ x, const bf16_t* __restrict__ w1,
             const float* __restrict__ b1, bf16_t* __restrict__ xf) {
  __shared__ __align__(16) unsigned ldsU[64 * CSTRIDE];
  const int b = blockIdx.z;
  const int n0 = blockIdx.x * 64;
  const int o0 = blockIdx.y * 128;
  const int t = threadIdx.x;
  const int wave = t >> 5, lane = t & 31, lo = lane & 15, hi = lane >> 4;
  const float* xb = x + (size_t)b * CC * NN;
  const int orow = o0 + wave * 16 + lo;

  v8f acc[4] = {{}, {}, {}, {}};
  for (int cc = 0; cc < CC; cc += 32) {
    __syncthreads();
    {  // stage X[cc..cc+31, n0..n0+63] column-major, K-pairs packed
      int rp = t >> 4, cg = t & 15;
      const float* s0 = xb + (size_t)(cc + 2 * rp) * NN + n0 + cg * 4;
      const float* s1 = s0 + NN;
      if (cc + 32 < CC) __builtin_prefetch(s0 + 32 * NN, 0, 1);
      float4 r0 = *(const float4*)s0, r1 = *(const float4*)s1;
      unsigned* d = ldsU + (cg * 4) * CSTRIDE + rp;
      d[0 * CSTRIDE] = pack2bf(r0.x, r1.x);
      d[1 * CSTRIDE] = pack2bf(r0.y, r1.y);
      d[2 * CSTRIDE] = pack2bf(r0.z, r1.z);
      d[3 * CSTRIDE] = pack2bf(r0.w, r1.w);
    }
    __syncthreads();
    v16bf a = frag_global(w1 + (size_t)orow * CC + cc + hi * 16);
    v16bf bf[4];  // preload all B fragments, then back-to-back WMMAs
    #pragma unroll
    for (int nt = 0; nt < 4; ++nt) bf[nt] = frag_lds(ldsU, nt * 16 + lo, hi);
    #pragma unroll
    for (int nt = 0; nt < 4; ++nt) acc[nt] = wmma_bf16(a, bf[nt], acc[nt]);
  }
  #pragma unroll
  for (int nt = 0; nt < 4; ++nt) {
    int col = n0 + nt * 16 + lo;
    #pragma unroll
    for (int i = 0; i < 8; ++i) {
      int row = o0 + wave * 16 + hi * 8 + i;
      float v = acc[nt][i] + b1[row];
      xf[(size_t)b * CC * NN + (size_t)row * NN + col] = (bf16_t)v;
    }
  }
}

// ---------------------------------------------------------------- z (E-step)
// S[n,k] = sum_c xf[c,n]*mu[c,k]; z = softmax_k(S); colsum[k] += sum_n z
// grid (NN/64, 1, BB), block 128 (4 waves); wave = 16(n) x 64(k)
__global__ void __launch_bounds__(128)
z_kernel(const bf16_t* __restrict__ xf, const bf16_t* __restrict__ mu,
         float* __restrict__ z, float* __restrict__ colsum) {
  __shared__ __align__(16) unsigned ldsXU[64 * CSTRIDE];  // cols = n, K = c
  __shared__ __align__(16) unsigned ldsMU[64 * CSTRIDE];  // cols = k, K = c
  __shared__ __align__(16) float    ldsS[64 * 64];
  const int b = blockIdx.z;
  const int n0 = blockIdx.x * 64;
  const int t = threadIdx.x;
  const int wave = t >> 5, lane = t & 31, lo = lane & 15, hi = lane >> 4;
  const bf16_t* xb = xf + (size_t)b * CC * NN;
  const bf16_t* mb = mu + (size_t)b * CC * KK;

  v8f acc[4] = {{}, {}, {}, {}};
  for (int cc = 0; cc < CC; cc += 32) {
    __syncthreads();
    {
      int rp = t >> 3, cg = t & 7;  // rp: c-pair 0..15, cg: 8 cols each
      const bf16_t* sx0 = xb + (size_t)(cc + 2 * rp) * NN + n0 + cg * 8;
      const bf16_t* sm0 = mb + (size_t)(cc + 2 * rp) * KK + cg * 8;
      if (cc + 32 < CC) __builtin_prefetch(sx0 + 32 * NN, 0, 1);
      union { int4 q; unsigned short h[8]; } x0, x1, m0, m1;
      x0.q = *(const int4*)sx0;  x1.q = *(const int4*)(sx0 + NN);
      m0.q = *(const int4*)sm0;  m1.q = *(const int4*)(sm0 + KK);
      #pragma unroll
      for (int j = 0; j < 8; ++j) {
        ldsXU[(cg * 8 + j) * CSTRIDE + rp] = (unsigned)x0.h[j] | ((unsigned)x1.h[j] << 16);
        ldsMU[(cg * 8 + j) * CSTRIDE + rp] = (unsigned)m0.h[j] | ((unsigned)m1.h[j] << 16);
      }
    }
    __syncthreads();
    // A = xf^T (row n, K=c): column-major-by-n layout IS the transpose
    v16bf a = frag_lds(ldsXU, wave * 16 + lo, hi);
    v16bf bf[4];
    #pragma unroll
    for (int kt = 0; kt < 4; ++kt) bf[kt] = frag_lds(ldsMU, kt * 16 + lo, hi);
    #pragma unroll
    for (int kt = 0; kt < 4; ++kt) acc[kt] = wmma_bf16(a, bf[kt], acc[kt]);
  }
  __syncthreads();
  #pragma unroll
  for (int kt = 0; kt < 4; ++kt)
    #pragma unroll
    for (int i = 0; i < 8; ++i)
      ldsS[(wave * 16 + hi * 8 + i) * 64 + kt * 16 + lo] = acc[kt][i];
  __syncthreads();
  if (t < 64) {  // softmax over 64 k-values, one row per thread
    float* row = ldsS + t * 64;
    float m = row[0];
    #pragma unroll
    for (int k = 1; k < 64; ++k) m = fmaxf(m, row[k]);
    float s = 0.f;
    #pragma unroll
    for (int k = 0; k < 64; ++k) { float e = __expf(row[k] - m); row[k] = e; s += e; }
    float inv = 1.f / s;
    float* zr = z + (size_t)b * NN * KK + (size_t)(n0 + t) * KK;
    #pragma unroll
    for (int k = 0; k < 64; ++k) { float v = row[k] * inv; row[k] = v; zr[k] = v; }
  }
  __syncthreads();
  if (t < 64) {  // per-k partial column sum, one atomic per (block,k)
    float s = 0.f;
    #pragma unroll
    for (int r = 0; r < 64; ++r) s += ldsS[r * 64 + t];
    atomicAdd(&colsum[b * KK + t], s);
  }
}

// ---------------------------------------------------------------- mu (M-step GEMM)
// mu_raw[c,k] = (sum_n xf[c,n]*z[n,k]) / (1e-6 + colsum[k])
// grid (CC/64, 1, BB), block 128 (4 waves); wave = 16(c) x 64(k)
__global__ void __launch_bounds__(128)
mu_gemm_kernel(const bf16_t* __restrict__ xf, const float* __restrict__ z,
               const float* __restrict__ colsum, float* __restrict__ mu_raw) {
  __shared__ __align__(16) unsigned ldsU[64 * CSTRIDE];  // cols = k, K = n
  const int b = blockIdx.z;
  const int c0 = blockIdx.x * 64;
  const int t = threadIdx.x;
  const int wave = t >> 5, lane = t & 31, lo = lane & 15, hi = lane >> 4;
  const bf16_t* xb = xf + (size_t)b * CC * NN;
  const float* zb = z + (size_t)b * NN * KK;

  v8f acc[4] = {{}, {}, {}, {}};
  for (int nn = 0; nn < NN; nn += 32) {
    __syncthreads();
    {
      int rp = t >> 3, cg = t & 7;  // rp: n-pair, cg: 8 k-cols
      const float* s0 = zb + (size_t)(nn + 2 * rp) * KK + cg * 8;
      const float* s1 = s0 + KK;
      if (nn + 32 < NN) __builtin_prefetch(s0 + 32 * KK, 0, 1);
      float4 r00 = ((const float4*)s0)[0], r01 = ((const float4*)s0)[1];
      float4 r10 = ((const float4*)s1)[0], r11 = ((const float4*)s1)[1];
      unsigned* d = ldsU + (cg * 8) * CSTRIDE + rp;
      d[0 * CSTRIDE] = pack2bf(r00.x, r10.x);
      d[1 * CSTRIDE] = pack2bf(r00.y, r10.y);
      d[2 * CSTRIDE] = pack2bf(r00.z, r10.z);
      d[3 * CSTRIDE] = pack2bf(r00.w, r10.w);
      d[4 * CSTRIDE] = pack2bf(r01.x, r11.x);
      d[5 * CSTRIDE] = pack2bf(r01.y, r11.y);
      d[6 * CSTRIDE] = pack2bf(r01.z, r11.z);
      d[7 * CSTRIDE] = pack2bf(r01.w, r11.w);
    }
    __syncthreads();
    v16bf a = frag_global(xb + (size_t)(c0 + wave * 16 + lo) * NN + nn + hi * 16);
    v16bf bf[4];
    #pragma unroll
    for (int kt = 0; kt < 4; ++kt) bf[kt] = frag_lds(ldsU, kt * 16 + lo, hi);
    #pragma unroll
    for (int kt = 0; kt < 4; ++kt) acc[kt] = wmma_bf16(a, bf[kt], acc[kt]);
  }
  #pragma unroll
  for (int kt = 0; kt < 4; ++kt) {
    int k = kt * 16 + lo;
    float inv = 1.f / (1e-6f + colsum[b * KK + k]);
    #pragma unroll
    for (int i = 0; i < 8; ++i) {
      int row = c0 + wave * 16 + hi * 8 + i;
      mu_raw[(size_t)b * CC * KK + (size_t)row * KK + k] = acc[kt][i] * inv;
    }
  }
}

// l2-normalize mu over c, emit bf16 copy. grid (KK, BB), block 128.
__global__ void __launch_bounds__(128)
mu_norm_kernel(const float* __restrict__ mu_raw, bf16_t* __restrict__ mub) {
  __shared__ float red[128];
  const int b = blockIdx.y, k = blockIdx.x;
  const int t = threadIdx.x;
  const float* src = mu_raw + (size_t)b * CC * KK;
  float s = 0.f;
  for (int c = t; c < CC; c += 128) { float v = src[(size_t)c * KK + k]; s += v * v; }
  red[t] = s; __syncthreads();
  for (int off = 64; off; off >>= 1) {
    if (t < off) red[t] += red[t + off];
    __syncthreads();
  }
  float scale = 1.f / (1e-6f + sqrtf(red[0]));
  for (int c = t; c < CC; c += 128)
    mub[(size_t)b * CC * KK + (size_t)c * KK + k] = (bf16_t)(src[(size_t)c * KK + k] * scale);
}

// ---------------------------------------------------------------- recon
// recon[c,n] = relu( sum_k mu[c,k]*z[n,k] ), K=64 -> 2 WMMA steps, no LDS
// grid (NN/64, CC/128, BB), block 256 (8 waves); wave = 16(c) x 64(n)
__global__ void __launch_bounds__(256)
recon_kernel(const bf16_t* __restrict__ mub, const float* __restrict__ z,
             bf16_t* __restrict__ recon) {
  const int b = blockIdx.z;
  const int n0 = blockIdx.x * 64;
  const int c0 = blockIdx.y * 128;
  const int t = threadIdx.x;
  const int wave = t >> 5, lane = t & 31, lo = lane & 15, hi = lane >> 4;
  const bf16_t* mb = mub + (size_t)b * CC * KK;
  const float* zb = z + (size_t)b * NN * KK;
  const int crow = c0 + wave * 16 + lo;

  v8f acc[4] = {{}, {}, {}, {}};
  #pragma unroll
  for (int kkp = 0; kkp < 2; ++kkp) {
    const int kk = kkp * 32;
    v16bf a = frag_global(mb + (size_t)crow * KK + kk + hi * 16);
    v16bf bf[4];
    #pragma unroll
    for (int nt = 0; nt < 4; ++nt) {
      // B = z^T: col n fixed per lane, 16 contiguous k floats -> 4x b128 + pk-cvt
      const float* bp = zb + (size_t)(n0 + nt * 16 + lo) * KK + kk + hi * 16;
      float4 f0 = ((const float4*)bp)[0], f1 = ((const float4*)bp)[1];
      float4 f2 = ((const float4*)bp)[2], f3 = ((const float4*)bp)[3];
      union { v16bf v; unsigned u[8]; } fb;
      fb.u[0] = pack2bf(f0.x, f0.y); fb.u[1] = pack2bf(f0.z, f0.w);
      fb.u[2] = pack2bf(f1.x, f1.y); fb.u[3] = pack2bf(f1.z, f1.w);
      fb.u[4] = pack2bf(f2.x, f2.y); fb.u[5] = pack2bf(f2.z, f2.w);
      fb.u[6] = pack2bf(f3.x, f3.y); fb.u[7] = pack2bf(f3.z, f3.w);
      bf[nt] = fb.v;
    }
    #pragma unroll
    for (int nt = 0; nt < 4; ++nt) acc[nt] = wmma_bf16(a, bf[nt], acc[nt]);
  }
  #pragma unroll
  for (int nt = 0; nt < 4; ++nt) {
    int col = n0 + nt * 16 + lo;
    #pragma unroll
    for (int i = 0; i < 8; ++i) {
      int row = c0 + wave * 16 + hi * 8 + i;
      float v = fmaxf(acc[nt][i], 0.f);
      recon[(size_t)b * CC * NN + (size_t)row * NN + col] = (bf16_t)v;
    }
  }
}

// ---------------------------------------------------------------- conv2
// y[b,o,n] = sum_c w2[o,c]*recon[c,n] (fp32 out for BN)
// grid (NN/64, CC/128, BB), block 256 (8 waves)
__global__ void __launch_bounds__(256)
conv2_kernel(const bf16_t* __restrict__ recon, const bf16_t* __restrict__ w2,
             float* __restrict__ y) {
  __shared__ __align__(16) unsigned ldsU[64 * CSTRIDE];
  const int b = blockIdx.z;
  const int n0 = blockIdx.x * 64, o0 = blockIdx.y * 128;
  const int t = threadIdx.x;
  const int wave = t >> 5, lane = t & 31, lo = lane & 15, hi = lane >> 4;
  const bf16_t* rb = recon + (size_t)b * CC * NN;
  const int orow = o0 + wave * 16 + lo;

  v8f acc[4] = {{}, {}, {}, {}};
  for (int cc = 0; cc < CC; cc += 32) {
    __syncthreads();
    {
      int rp = t >> 4, cg = t & 15;
      const bf16_t* s0 = rb + (size_t)(cc + 2 * rp) * NN + n0 + cg * 4;
      if (cc + 32 < CC) __builtin_prefetch(s0 + 32 * NN, 0, 1);
      union { uint2 q; unsigned short h[4]; } a0, a1;
      a0.q = *(const uint2*)s0;
      a1.q = *(const uint2*)(s0 + NN);
      unsigned* d = ldsU + (cg * 4) * CSTRIDE + rp;
      d[0 * CSTRIDE] = (unsigned)a0.h[0] | ((unsigned)a1.h[0] << 16);
      d[1 * CSTRIDE] = (unsigned)a0.h[1] | ((unsigned)a1.h[1] << 16);
      d[2 * CSTRIDE] = (unsigned)a0.h[2] | ((unsigned)a1.h[2] << 16);
      d[3 * CSTRIDE] = (unsigned)a0.h[3] | ((unsigned)a1.h[3] << 16);
    }
    __syncthreads();
    v16bf a = frag_global(w2 + (size_t)orow * CC + cc + hi * 16);
    v16bf bf[4];
    #pragma unroll
    for (int nt = 0; nt < 4; ++nt) bf[nt] = frag_lds(ldsU, nt * 16 + lo, hi);
    #pragma unroll
    for (int nt = 0; nt < 4; ++nt) acc[nt] = wmma_bf16(a, bf[nt], acc[nt]);
  }
  #pragma unroll
  for (int nt = 0; nt < 4; ++nt) {
    int col = n0 + nt * 16 + lo;
    #pragma unroll
    for (int i = 0; i < 8; ++i) {
      int row = o0 + wave * 16 + hi * 8 + i;
      y[(size_t)b * CC * NN + (size_t)row * NN + col] = acc[nt][i];
    }
  }
}

// ---------------------------------------------------------------- BatchNorm
__global__ void __launch_bounds__(256)
bn_stats_kernel(const float* __restrict__ y, float* __restrict__ stat) {
  __shared__ float rs[256], rs2[256];
  const int o = blockIdx.x, t = threadIdx.x;
  float s = 0.f, s2 = 0.f;
  for (int b = 0; b < BB; ++b) {
    const float* p = y + ((size_t)b * CC + o) * NN;
    for (int j = t * 4; j < NN; j += 1024) {
      float4 v = *(const float4*)(p + j);
      s += v.x + v.y + v.z + v.w;
      s2 += v.x * v.x + v.y * v.y + v.z * v.z + v.w * v.w;
    }
  }
  rs[t] = s; rs2[t] = s2; __syncthreads();
  for (int off = 128; off; off >>= 1) {
    if (t < off) { rs[t] += rs[t + off]; rs2[t] += rs2[t + off]; }
    __syncthreads();
  }
  if (t == 0) { stat[o] = rs[0]; stat[CC + o] = rs2[0]; }
}

__global__ void bn_finalize_kernel(const float* __restrict__ stat,
                                   const float* __restrict__ gamma,
                                   const float* __restrict__ beta,
                                   float* __restrict__ scale,
                                   float* __restrict__ shift) {
  int o = blockIdx.x * 256 + threadIdx.x;
  if (o < CC) {
    const float cnt = (float)(BB * NN);
    float mean = stat[o] / cnt;
    float var = stat[CC + o] / cnt - mean * mean;  // biased var
    float sc = gamma[o] * rsqrtf(var + BN_EPS_F);
    scale[o] = sc;
    shift[o] = beta[o] - mean * sc;
  }
}

// out = relu(y*scale[ch] + shift[ch] + x); float4 vectorized
__global__ void __launch_bounds__(256)
bn_out_kernel(const float* __restrict__ y, const float* __restrict__ x,
              const float* __restrict__ scale, const float* __restrict__ shift,
              float* __restrict__ out) {
  size_t i = ((size_t)blockIdx.x * 256 + threadIdx.x) * 4;
  int ch = (int)((i / NN) % CC);
  float4 yv = *(const float4*)(y + i);
  float4 xv = *(const float4*)(x + i);
  float sc = scale[ch], sh = shift[ch];
  float4 o;
  o.x = fmaxf(yv.x * sc + sh + xv.x, 0.f);
  o.y = fmaxf(yv.y * sc + sh + xv.y, 0.f);
  o.z = fmaxf(yv.z * sc + sh + xv.z, 0.f);
  o.w = fmaxf(yv.w * sc + sh + xv.w, 0.f);
  *(float4*)(out + i) = o;
}

// ---------------------------------------------------------------- launcher
extern "C" void kernel_launch(void* const* d_in, const int* in_sizes, int n_in,
                              void* d_out, int out_size, void* d_ws, size_t ws_size,
                              hipStream_t stream) {
  const float* x     = (const float*)d_in[0];
  const float* mu0   = (const float*)d_in[1];
  const float* w1    = (const float*)d_in[2];
  const float* b1    = (const float*)d_in[3];
  const float* w2    = (const float*)d_in[4];
  const float* gamma = (const float*)d_in[5];
  const float* beta  = (const float*)d_in[6];
  float* out = (float*)d_out;

  char* ws = (char*)d_ws;
  size_t off = 0;
  auto alloc = [&](size_t bytes) -> void* {
    void* p = ws + off;
    off += (bytes + 255) & ~(size_t)255;
    return p;
  };
  bf16_t* xf   = (bf16_t*)alloc((size_t)BB * CC * NN * sizeof(bf16_t));  // 64 MiB
  float*  z    = (float*) alloc((size_t)BB * NN * KK * sizeof(float));   // 16 MiB
  float*  murw = (float*) alloc((size_t)BB * CC * KK * sizeof(float));   //  2 MiB
  bf16_t* mub  = (bf16_t*)alloc((size_t)BB * CC * KK * sizeof(bf16_t));  //  1 MiB
  float*  cs   = (float*) alloc((size_t)BB * KK * sizeof(float));
  bf16_t* rec  = (bf16_t*)alloc((size_t)BB * CC * NN * sizeof(bf16_t));  // 64 MiB
  float*  y    = (float*) alloc((size_t)BB * CC * NN * sizeof(float));   // 128 MiB
  bf16_t* w1b  = (bf16_t*)alloc((size_t)CC * CC * sizeof(bf16_t));
  bf16_t* w2b  = (bf16_t*)alloc((size_t)CC * CC * sizeof(bf16_t));
  float*  stat = (float*) alloc(2 * CC * sizeof(float));
  float*  bnsc = (float*) alloc(CC * sizeof(float));
  float*  bnsh = (float*) alloc(CC * sizeof(float));

  cvt_weights_kernel<<<(CC * CC + 255) / 256, 256, 0, stream>>>(w1, w2, mu0, w1b, w2b, mub);
  conv1_kernel<<<dim3(NN / 64, CC / 128, BB), 256, 0, stream>>>(x, w1b, b1, xf);
  for (int s = 0; s < 3; ++s) {
    zero_kernel<<<(BB * KK + 255) / 256, 256, 0, stream>>>(cs, BB * KK);
    z_kernel<<<dim3(NN / 64, 1, BB), 128, 0, stream>>>(xf, mub, z, cs);
    mu_gemm_kernel<<<dim3(CC / 64, 1, BB), 128, 0, stream>>>(xf, z, cs, murw);
    mu_norm_kernel<<<dim3(KK, BB), 128, 0, stream>>>(murw, mub);
  }
  recon_kernel<<<dim3(NN / 64, CC / 128, BB), 256, 0, stream>>>(mub, z, rec);
  conv2_kernel<<<dim3(NN / 64, CC / 128, BB), 256, 0, stream>>>(rec, w2b, y);
  bn_stats_kernel<<<CC, 256, 0, stream>>>(y, stat);
  bn_finalize_kernel<<<(CC + 255) / 256, 256, 0, stream>>>(stat, gamma, beta, bnsc, bnsh);
  bn_out_kernel<<<(unsigned)((size_t)BB * CC * NN / 1024), 256, 0, stream>>>(y, x, bnsc, bnsh, out);
}
